// CNF_10694468567320
// MI455X (gfx1250) — matrix operand order repaired and checked by reading, not compiled, not run on hardware
//
#include <hip/hip_runtime.h>

typedef __bf16 bf16x16 __attribute__((ext_vector_type(16)));
typedef float  f32x8   __attribute__((ext_vector_type(8)));
typedef unsigned short u16;
typedef unsigned short u16x8 __attribute__((ext_vector_type(8)));

union FragBF {
    bf16x16 bf;
    u16x8   us[2];
};

#define HID 512
#define NKH 16       // K-tiles for hidden layers (512/32)
#define NK1 3        // K-tiles for layer1 (96/32, padded from 81)
#define IN_PAD 96
#define MT 4         // row-tiles of 16 -> 64 samples per workgroup
#define MROWS (16 * MT)

// Tsit5 tableau in constant memory (scalar loads, uniform across wave)
__constant__ float d_AT[6][5] = {
    {0.f, 0.f, 0.f, 0.f, 0.f},
    {0.161f, 0.f, 0.f, 0.f, 0.f},
    {-0.008480655492356989f, 0.335480655492357f, 0.f, 0.f, 0.f},
    {2.8971530571054935f, -6.359448489975075f, 4.3622954328695815f, 0.f, 0.f},
    {5.325864828439257f, -11.748883564062828f, 7.4955393428898365f, -0.09249506636175525f, 0.f},
    {5.86145544294642f, -12.92096931784711f, 8.159367898576159f, -0.071584973281401f, -0.028269050394068383f}};
__constant__ float d_CT[6] = {0.0f, 0.161f, 0.327f, 0.9f, 0.9800255409045097f, 1.0f};
__constant__ float d_BT[6] = {0.09646076681806523f, 0.01f, 0.4798896504144996f,
                              1.379008574103742f, -3.290069515436081f, 2.324710524099774f};

__device__ __forceinline__ u16 f2bf(float f) {
    union { float f; unsigned u; } v; v.f = f;
    unsigned r = v.u + 0x7FFFu + ((v.u >> 16) & 1u);   // round-to-nearest-even
    return (u16)(r >> 16);
}

__device__ __forceinline__ float gelu_tanh(float v) {
    // 0.5*v*(1+tanh(0.79788456*(v + 0.044715 v^3))) with hw v_tanh_f32
    float v2 = v * v;
    float c  = v * fmaf(v2, 0.035677408136300125f, 0.7978845608028654f);
    float th;
#if __has_builtin(__builtin_amdgcn_tanhf)
    th = __builtin_amdgcn_tanhf(c);
#else
    th = tanhf(c);
#endif
    float hv = 0.5f * v;
    return fmaf(hv, th, hv);
}

// ---------------------------------------------------------------------------
// Repack fp32 weight [K][N] (row-major) into bf16 WMMA B-fragment order:
// [ntile][ktile][lane 0..31][16 contiguous bf16]
//   lane holds column n = ntile*16 + lane%16,
//   K run:  kt*32 + (lane<16 ? 0 : 8) + j + (j<8 ? 0 : 8) , j=0..15
// ---------------------------------------------------------------------------
__global__ void repack_weights(const float* __restrict__ W, u16* __restrict__ Wp,
                               int K, int N, int nkTiles) {
    int idx = blockIdx.x * blockDim.x + threadIdx.x;
    int total = (N / 16) * nkTiles * 32 * 16;
    if (idx >= total) return;
    int j    = idx & 15;
    int lane = (idx >> 4) & 31;
    int kt   = (idx >> 9) % nkTiles;
    int nt   = (idx >> 9) / nkTiles;
    int k = kt * 32 + ((lane < 16) ? 0 : 8) + j + ((j < 8) ? 0 : 8);
    int n = nt * 16 + (lane & 15);
    float v = (k < K) ? W[k * N + n] : 0.0f;
    Wp[idx] = f2bf(v);
}

// ---------------------------------------------------------------------------
// Hidden layer: [64 x Kin] (bf16, LDS) @ [Kin x 512] (bf16 packed, L2)
// -> gelu -> [64 x 512] bf16 in LDS.
// Each wave owns 4 N-tiles x 4 M-tiles, processed as TWO sequential passes of
// 2 N-tiles so only 64 accumulator VGPRs are live at a time (no spills),
// while every B fragment still feeds 4 WMMAs (4x weight-bandwidth reuse).
// ---------------------------------------------------------------------------
__device__ __forceinline__ void hidden_layer(const u16* inL, int strideIn, int nk,
                                             const u16* __restrict__ Wp,
                                             const float* biasL,
                                             u16* outL, int lane, int wave) {
    const int rlo   = lane & 15;
    const int koffA = (lane < 16) ? 0 : 8;    // A-matrix 16-bit layout (ISA 7.12.2)
    const int rhalf = (lane < 16) ? 0 : 8;    // C/D layout: VGPR v -> row rhalf+v

#pragma unroll 1
    for (int half = 0; half < 2; ++half) {
        f32x8 acc[2][MT];
#pragma unroll
        for (int nt = 0; nt < 2; ++nt)
#pragma unroll
            for (int mt = 0; mt < MT; ++mt)
                acc[nt][mt] = (f32x8){0.f,0.f,0.f,0.f,0.f,0.f,0.f,0.f};

#pragma unroll 1
        for (int kt = 0; kt < nk; ++kt) {
            FragBF a[MT];
#pragma unroll
            for (int mt = 0; mt < MT; ++mt) {
                const u16x8* pa =
                    (const u16x8*)&inL[(mt * 16 + rlo) * strideIn + kt * 32 + koffA];
                a[mt].us[0] = pa[0];      // K = kbase .. kbase+7
                a[mt].us[1] = pa[2];      // K = kbase+16 .. kbase+23
            }
#pragma unroll
            for (int nt = 0; nt < 2; ++nt) {
                const int ntg = wave * 4 + half * 2 + nt;
                const u16x8* pb = (const u16x8*)&Wp[(((ntg * nk) + kt) * 32 + lane) * 16];
                FragBF b;
                b.us[0] = pb[0];
                b.us[1] = pb[1];
#pragma unroll
                for (int mt = 0; mt < MT; ++mt)
                    acc[nt][mt] = __builtin_amdgcn_wmma_f32_16x16x32_bf16(
                        false, a[mt].bf, false, b.bf, (short)0, acc[nt][mt], false, false);
            }
        }

#pragma unroll
        for (int nt = 0; nt < 2; ++nt) {
            const int col = (wave * 4 + half * 2 + nt) * 16 + rlo;
            const float bs = biasL[col];
#pragma unroll
            for (int mt = 0; mt < MT; ++mt) {
#pragma unroll
                for (int v = 0; v < 8; ++v) {
                    float xv = acc[nt][mt][v] + bs;
                    outL[(mt * 16 + rhalf + v) * HID + col] = f2bf(gelu_tanh(xv));
                }
            }
        }
    }
}

// Output layer: [64 x 512] @ [512 x 16]; K split across 8 waves (2 K-tiles each),
// partials reduced into LDS f32 via ds_add_f32 atomics.
__device__ __forceinline__ void out_layer(const u16* inL,
                                          const u16* __restrict__ Wop,
                                          float* scoreBuf, int lane, int wave) {
    f32x8 acc[MT];
#pragma unroll
    for (int mt = 0; mt < MT; ++mt) acc[mt] = (f32x8){0.f,0.f,0.f,0.f,0.f,0.f,0.f,0.f};
    const int rlo   = lane & 15;
    const int koffA = (lane < 16) ? 0 : 8;
#pragma unroll
    for (int i = 0; i < 2; ++i) {
        const int kt = wave * 2 + i;
        FragBF b;
        const u16x8* pb = (const u16x8*)&Wop[(kt * 32 + lane) * 16];
        b.us[0] = pb[0];
        b.us[1] = pb[1];
#pragma unroll
        for (int mt = 0; mt < MT; ++mt) {
            FragBF a;
            const u16x8* pa = (const u16x8*)&inL[(mt * 16 + rlo) * HID + kt * 32 + koffA];
            a.us[0] = pa[0];
            a.us[1] = pa[2];
            acc[mt] = __builtin_amdgcn_wmma_f32_16x16x32_bf16(
                false, a.bf, false, b.bf, (short)0, acc[mt], false, false);
        }
    }
    const int rhalf = (lane < 16) ? 0 : 8;
#pragma unroll
    for (int mt = 0; mt < MT; ++mt)
#pragma unroll
        for (int v = 0; v < 8; ++v)
            atomicAdd(&scoreBuf[(mt * 16 + rhalf + v) * 16 + rlo], acc[mt][v]);
}

// ---------------------------------------------------------------------------
// Persistent per-tile ODE integrator: each block owns 64 samples and runs the
// full 100-step Tsit5 integration (600 MLP evals) entirely on-chip.
// ---------------------------------------------------------------------------
__global__ __launch_bounds__(256)
void cnf_tsit5_kernel(const float* __restrict__ x,
                      const float* __restrict__ theta0,
                      const float* __restrict__ b1,
                      const float* __restrict__ b2,
                      const float* __restrict__ b3,
                      const float* __restrict__ bout,
                      const float* __restrict__ pmean,
                      const float* __restrict__ pstd,
                      const float* __restrict__ dmean,
                      const float* __restrict__ dstd,
                      const u16* __restrict__ W1p,
                      const u16* __restrict__ W2p,
                      const u16* __restrict__ W3p,
                      const u16* __restrict__ Wop,
                      float* __restrict__ out) {
    __shared__ alignas(16) u16 A1[MROWS * IN_PAD];    // 12 KB layer-1 input
    __shared__ alignas(16) u16 hA[MROWS * HID];       // 64 KB activations
    __shared__ alignas(16) u16 hB[MROWS * HID];       // 64 KB activations
    __shared__ float scoreBuf[MROWS * 16];            // 4 KB score reduce
    __shared__ float kbuf[6][MROWS * 16];             // 24 KB Tsit5 k-stages
    __shared__ float bl1[HID], bl2[HID], bl3[HID];    // 6 KB biases
    __shared__ float boutL[16];
    __shared__ float xn[64];

    const int tid  = threadIdx.x;
    const int lane = tid & 31;
    const int wave = tid >> 5;
    const int s    = tid >> 4;     // sample-in-tile row base 0..15
    const int p    = tid & 15;     // parameter 0..15

    // normalize conditioning data + stage biases once
    if (tid < 64) xn[tid] = (x[tid] - dmean[tid]) / dstd[tid];
    for (int i = tid; i < HID; i += 256) {
        bl1[i] = b1[i];
        bl2[i] = b2[i];
        bl3[i] = b3[i];
    }
    if (tid < 16) boutL[tid] = bout[tid];
    __syncthreads();

    // x-block of layer-1 input (constant across all evals): cols 16..79
    for (int i = tid; i < MROWS * 64; i += 256) {
        int r = i >> 6, c = i & 63;
        A1[r * IN_PAD + 16 + c] = f2bf(xn[c]);
    }
    // zero pad: cols 81..95
    for (int i = tid; i < MROWS * 15; i += 256) {
        int r = i / 15, c = 81 + i % 15;
        A1[r * IN_PAD + c] = 0;
    }

    float y[MT];
#pragma unroll
    for (int e = 0; e < MT; ++e)
        y[e] = theta0[(blockIdx.x * MROWS + s + 16 * e) * 16 + p];

    const float dt = (1e-5f - 1.0f) / 100.0f;

#pragma unroll 1
    for (int step = 0; step < 100; ++step) {
        const float t = 1.0f + dt * (float)step;
#pragma unroll 1
        for (int st = 0; st < 6; ++st) {           // dynamic: MLP body appears once
            const float tst = t + d_CT[st] * dt;
            float ys[MT];
#pragma unroll
            for (int e = 0; e < MT; ++e) {
                const int row = s + 16 * e;
                float v = y[e];
                for (int j = 0; j < st; ++j)
                    v = fmaf(dt * d_AT[st][j], kbuf[j][row * 16 + p], v);
                ys[e] = v;
                A1[row * IN_PAD + p] = f2bf(v);
                scoreBuf[row * 16 + p] = boutL[p];
            }
            if (p == 0) {
#pragma unroll
                for (int e = 0; e < MT; ++e)
                    A1[(s + 16 * e) * IN_PAD + 80] = f2bf(tst);
            }
            __syncthreads();

            hidden_layer(A1, IN_PAD, NK1, W1p, bl1, hA, lane, wave);
            __syncthreads();
            hidden_layer(hA, HID, NKH, W2p, bl2, hB, lane, wave);
            __syncthreads();
            hidden_layer(hB, HID, NKH, W3p, bl3, hA, lane, wave);
            __syncthreads();
            out_layer(hA, Wop, scoreBuf, lane, wave);
            __syncthreads();

            // drift: k = -0.5*beta(t)*(y_stage + score)
            const float beta = 0.1f + tst * 19.9f;
#pragma unroll
            for (int e = 0; e < MT; ++e) {
                const int row = s + 16 * e;
                kbuf[st][row * 16 + p] = -0.5f * beta * (ys[e] + scoreBuf[row * 16 + p]);
            }
            __syncthreads();
        }
#pragma unroll
        for (int e = 0; e < MT; ++e) {
            const int row = s + 16 * e;
            float acc = y[e];
#pragma unroll
            for (int j = 0; j < 6; ++j)
                acc = fmaf(dt * d_BT[j], kbuf[j][row * 16 + p], acc);
            y[e] = acc;
        }
    }

#pragma unroll
    for (int e = 0; e < MT; ++e) {
        const int gs = blockIdx.x * MROWS + s + 16 * e;
        out[gs * 16 + p] = pmean[p] + pstd[p] * y[e];
    }
}

extern "C" void kernel_launch(void* const* d_in, const int* in_sizes, int n_in,
                              void* d_out, int out_size, void* d_ws, size_t ws_size,
                              hipStream_t stream) {
    const float* x      = (const float*)d_in[0];
    const float* theta0 = (const float*)d_in[1];
    const float* W1     = (const float*)d_in[2];
    const float* b1     = (const float*)d_in[3];
    const float* W2     = (const float*)d_in[4];
    const float* b2     = (const float*)d_in[5];
    const float* W3     = (const float*)d_in[6];
    const float* b3     = (const float*)d_in[7];
    const float* Wout   = (const float*)d_in[8];
    const float* bout   = (const float*)d_in[9];
    const float* pmean  = (const float*)d_in[10];
    const float* pstd   = (const float*)d_in[11];
    const float* dmean  = (const float*)d_in[12];
    const float* dstd   = (const float*)d_in[13];

    u16* ws  = (u16*)d_ws;
    u16* W1p = ws;                 // 32 ntiles * 3 kt * 512  = 49152
    u16* W2p = W1p + 49152;        // 32 * 16 * 512           = 262144
    u16* W3p = W2p + 262144;
    u16* Wop = W3p + 262144;       // 1 * 16 * 512            = 8192

    repack_weights<<<(49152 + 255) / 256, 256, 0, stream>>>(W1, W1p, 81, 512, NK1);
    repack_weights<<<(262144 + 255) / 256, 256, 0, stream>>>(W2, W2p, 512, 512, NKH);
    repack_weights<<<(262144 + 255) / 256, 256, 0, stream>>>(W3, W3p, 512, 512, NKH);
    repack_weights<<<(8192 + 255) / 256, 256, 0, stream>>>(Wout, Wop, 512, 16, NKH);

    cnf_tsit5_kernel<<<32768 / MROWS, 256, 0, stream>>>(
        x, theta0, b1, b2, b3, bout, pmean, pstd, dmean, dstd,
        W1p, W2p, W3p, Wop, (float*)d_out);
}
